// ConflictNet_91199335563636
// MI455X (gfx1250) — matrix operand order repaired and checked by reading, not compiled, not run on hardware
//
#include <hip/hip_runtime.h>
#include <hip/hip_bf16.h>

typedef __attribute__((ext_vector_type(16))) _Float16 v16h;
typedef __attribute__((ext_vector_type(8)))  float    v8f;

#define DEVFN __device__ __forceinline__

constexpr int   Bc = 2, Nc = 8000, DYn = 12, STDn = 6, Tc = 30;
constexpr int   ROWS = Bc * Nc;                  // 16000
constexpr float DXf = 0.5f, DYGf = 0.5f, DTf = 1.0f;

// f16 weight pool offsets (in halfs). Each matrix is stored in WMMA-B
// *fragment order*: frag = nt*Kc + kc, then [lane][16 halfs] contiguous.
constexpr int O_TR0 = 0;                 // 64x64  (trans_W0 62x64, K-pad)
constexpr int O_TR1 = O_TR0 + 64 * 64;   // 64x32
constexpr int O_TR2 = O_TR1 + 64 * 32;   // 32x16  (col 0 = trans_W2[:,25])
constexpr int O_Q0  = O_TR2 + 32 * 16;   // 64x64  (q_W0 38x64, K-pad)
constexpr int O_Q1  = O_Q0  + 64 * 64;   // 64x32
constexpr int O_Q2  = O_Q1  + 64 * 32;   // 32x16  (cols 0..11 = q_W2)
constexpr int O_D0  = O_Q2  + 32 * 16;   // 32x32  (dec_W0 18x32, K-pad)
constexpr int O_D1  = O_D0  + 32 * 32;   // 32x32
constexpr int O_D2  = O_D1  + 32 * 32;   // 32x64
constexpr int O_D3  = O_D2  + 32 * 64;   // 64x16  (col 0 = dec_W3)
constexpr int W_TOTAL = O_D3 + 64 * 16;  // 18432 halfs

// ---------------------------------------------------------------------------
// K0a: parameter processing (coeffM/Dij/MUij/v) + f16 weight conversion into
//      WMMA-B fragment layout (one contiguous v16h per lane per fragment).
// ---------------------------------------------------------------------------
__global__ void k_param(const float* __restrict__ D, const float* __restrict__ mu,
                        const float* __restrict__ cv,
                        const float* __restrict__ tw0, const float* __restrict__ tw1,
                        const float* __restrict__ tw2,
                        const float* __restrict__ qw0, const float* __restrict__ qw1,
                        const float* __restrict__ qw2,
                        const float* __restrict__ dw0, const float* __restrict__ dw1,
                        const float* __restrict__ dw2, const float* __restrict__ dw3,
                        float* __restrict__ params, _Float16* __restrict__ wp) {
  const int tid = threadIdx.x, bs = blockDim.x;
  if (tid < 144) {
    int r = tid / 12, c = tid % 12;
    int br = r / 4, bc = c / 4, lr = r % 4, lc = c % 4;
    auto diagv = [&](int i) -> float {
      int j = lr * 4 + lc;                       // 4x4 flat; diag ones at j%5==0
      return (j % 5 == 0) ? 1.0f : cv[64 + 12 * i + (j / 5) * 4 + (j % 5) - 1];
    };
    float val = 0.0f;
    if (br == 0)      { if (bc == 0) val = diagv(0); else if (bc == 2) val = cv[0  + lr * 4 + lc]; }
    else if (br == 1) { if (bc == 1) val = diagv(1); else if (bc == 2) val = cv[16 + lr * 4 + lc]; }
    else              { if (bc == 0) val = cv[32 + lr * 4 + lc];
                        else if (bc == 1) val = cv[48 + lr * 4 + lc];
                        else val = diagv(2); }
    params[tid] = fmaxf(val, 0.0f);                                   // coeffM
  } else if (tid < 156) {
    int c = tid - 144;
    params[144 + c] = (c < 4) ? 0.0f : fmaxf(D[c - 4], 0.0f);         // Dij
  } else if (tid < 168) {
    int c = tid - 156;
    params[156 + c] = (c < 4) ? fmaxf(mu[c], 0.0f)
                              : ((c < 8) ? 0.0f : fmaxf(mu[c - 4], 0.0f)); // MUij
  }
  __syncthreads();
  if (tid < 12) {                                  // v[d] = sum_c MUij[c]*coeffM[c][d]
    float s = 0.0f;
    for (int c = 0; c < 12; ++c) s += params[156 + c] * params[c * 12 + tid];
    params[168 + tid] = s;
  }
  // ---- weight conversion to padded f16, WMMA-B fragment order ----
  auto conv = [&](int off, int Kdim, int Nout, auto getv) {
    int Kc = Kdim / 32, total = Kdim * Nout;
    for (int idx = tid; idx < total; idx += bs) {
      int frag = idx >> 9, rem = idx & 511, lane = rem >> 4, j = rem & 15;
      int kc = frag % Kc, nt = frag / Kc, hl = lane >> 4, m = lane & 15;
      int K = kc * 32 + 8 * hl + ((j < 8) ? j : j + 8), n = nt * 16 + m;
      wp[off + idx] = (_Float16)getv(K, n);
    }
  };
  conv(O_TR0, 64, 64, [&](int k, int n) { return (k < 62) ? tw0[k * 64 + n] : 0.0f; });
  conv(O_TR1, 64, 32, [&](int k, int n) { return tw1[k * 32 + n]; });
  conv(O_TR2, 32, 16, [&](int k, int n) { return (n == 0) ? tw2[k * 26 + 25] : 0.0f; });
  conv(O_Q0,  64, 64, [&](int k, int n) { return (k < 38) ? qw0[k * 64 + n] : 0.0f; });
  conv(O_Q1,  64, 32, [&](int k, int n) { return qw1[k * 32 + n]; });
  conv(O_Q2,  32, 16, [&](int k, int n) { return (n < 12) ? qw2[k * 12 + n] : 0.0f; });
  conv(O_D0,  32, 32, [&](int k, int n) { return (k < 18) ? dw0[k * 32 + n] : 0.0f; });
  conv(O_D1,  32, 32, [&](int k, int n) { return dw1[k * 32 + n]; });
  conv(O_D2,  32, 64, [&](int k, int n) { return dw2[k * 64 + n]; });
  conv(O_D3,  64, 16, [&](int k, int n) { return (n == 0) ? dw3[k] : 0.0f; });
}

// ---------------------------------------------------------------------------
// K0b: per-node constants (gpx/gpy/phi_ag/st_ag) + feat init
// ---------------------------------------------------------------------------
__global__ void k_node(const float* __restrict__ phi, const float* __restrict__ feat_dy,
                       const float* __restrict__ fst,
                       const int* __restrict__ adj4, const int* __restrict__ adj25,
                       float* __restrict__ featb, float* __restrict__ gpx,
                       float* __restrict__ gpy, float* __restrict__ phiag,
                       float* __restrict__ stag) {
  int row = blockIdx.x * blockDim.x + threadIdx.x;
  if (row >= ROWS) return;
  int b = row / Nc, base = b * Nc;
  for (int c = 0; c < 12; ++c) featb[(size_t)row * 12 + c] = feat_dy[(size_t)row * 12 + c];
  int i0 = adj4[row * 4 + 0], i1 = adj4[row * 4 + 1];
  int i2 = adj4[row * 4 + 2], i3 = adj4[row * 4 + 3];
  gpx[row] = (phi[base + i1] - phi[base + i0]) / (2.0f * DXf);
  gpy[row] = (phi[base + i3] - phi[base + i2]) / (2.0f * DYGf);
  float ps = 0.0f, ss[6] = {0, 0, 0, 0, 0, 0};
  for (int k = 0; k < 25; ++k) {
    int idx = base + adj25[row * 25 + k];
    float w = (k == 0) ? 1.0f : 0.01f;
    ps += w * phi[idx];
    for (int c = 0; c < 6; ++c) ss[c] += w * fst[(size_t)idx * 6 + c];
  }
  phiag[row * 2 + 0] = phi[row];
  phiag[row * 2 + 1] = ps;
  for (int c = 0; c < 6; ++c) {
    stag[(size_t)row * 12 + c]     = fst[(size_t)row * 6 + c];
    stag[(size_t)row * 12 + 6 + c] = ss[c];
  }
}

__global__ void k_zero(float* __restrict__ p, int n) {
  int i = blockIdx.x * blockDim.x + threadIdx.x;
  if (i < n) p[i] = 0.0f;
}

// store one 64-wide f32 row into A-fragment-ordered f16 buffer
// layout: [(tile*2 + kc)*32 + lane][16], lane = hl*16 + (row%16)
DEVFN void store_row_frags(_Float16* __restrict__ dst, int row, const float* vals) {
  int tile = row >> 4, m = row & 15;
#pragma unroll
  for (int kc = 0; kc < 2; ++kc)
#pragma unroll
    for (int hl = 0; hl < 2; ++hl) {
      v16h h;
#pragma unroll
      for (int j = 0; j < 16; ++j) {
        int K = kc * 32 + 8 * hl + ((j < 8) ? j : j + 8);
        h[j] = (_Float16)vals[K];
      }
      *(v16h*)(dst + (((size_t)tile * 2 + kc) * 32 + (hl * 16 + m)) * 16) = h;
    }
}

// ---------------------------------------------------------------------------
// K1: per-step stencil + u_diff + probability scatter + MLP input assembly
// ---------------------------------------------------------------------------
__global__ void k_step1(const float* __restrict__ featb, const float* __restrict__ gpx,
                        const float* __restrict__ gpy, const float* __restrict__ phiag,
                        const float* __restrict__ stag, const float* __restrict__ params,
                        const int* __restrict__ adj4, const int* __restrict__ adj25,
                        float* __restrict__ udiff, float* __restrict__ urig,
                        _Float16* __restrict__ atr, _Float16* __restrict__ aq) {
  __shared__ float sp[180];
  for (int i = threadIdx.x; i < 180; i += blockDim.x) sp[i] = params[i];
  __syncthreads();
  int row = blockIdx.x * blockDim.x + threadIdx.x;
  if (row >= ROWS) return;
  const float* cM  = sp;        // 12x12
  const float* Dij = sp + 144;
  const float* MU  = sp + 156;
  const float* vv  = sp + 168;
  int b = row / Nc, base = b * Nc;
  int n0 = base + adj4[row * 4 + 0], n1 = base + adj4[row * 4 + 1];
  int n2 = base + adj4[row * 4 + 2], n3 = base + adj4[row * 4 + 3];
  float g0 = gpx[n0], g1 = gpx[n1], g2 = gpy[n2], g3 = gpy[n3];
  float f[12], tmp[12], fsum[12];
  for (int c = 0; c < 12; ++c) {
    float fc = featb[(size_t)row * 12 + c]; f[c] = fc;
    float f0 = featb[(size_t)n0 * 12 + c], f1 = featb[(size_t)n1 * 12 + c];
    float f2 = featb[(size_t)n2 * 12 + c], f3 = featb[(size_t)n3 * 12 + c];
    float J  = (f0 + f1 - 2.0f * fc) / (DXf * DXf) + (f2 + f3 - 2.0f * fc) / (DYGf * DYGf);
    float Fd = (g1 * f1 - g0 * f0) / (2.0f * DXf) + (g3 * f3 - g2 * f2) / (2.0f * DYGf);
    tmp[c] = Dij[c] * J - MU[c] * Fd;
  }
  int a25[25];
  for (int k = 0; k < 25; ++k) a25[k] = base + adj25[row * 25 + k];
  for (int c = 0; c < 12; ++c) fsum[c] = 0.0f;
  for (int k = 0; k < 25; ++k) {
    float w = (k == 0) ? 1.0f : 0.01f;
    const float* fp = &featb[(size_t)a25[k] * 12];
    for (int c = 0; c < 12; ++c) fsum[c] += w * fp[c];
  }
  for (int d = 0; d < 12; ++d) {
    float s = 0.0f;
    for (int c = 0; c < 12; ++c) s += tmp[c] * cM[c * 12 + d];
    udiff[(size_t)row * 12 + d] = s;
  }
  // transition-probability scatter: ovl(t,g) = max(0, 1-|t-g|)
  float gx = gpx[row], gy = gpy[row];
  for (int d = 0; d < 12; ++d) {
    float mass = (f[d] == 0.0f) ? 1.0f : f[d];
    float ax = fminf(fmaxf(gx * vv[d] / mass, -2.0f), 2.0f) * DTf;
    float ay = fminf(fmaxf(gy * vv[d] / mass, -2.0f), 2.0f) * DTf;
    float ox[5], oy[5];
    for (int g = 0; g < 5; ++g) {
      ox[g] = fmaxf(0.0f, 1.0f - fabsf(ax - (float)g));
      oy[g] = fmaxf(0.0f, 1.0f - fabsf(ay - (float)g));
    }
    float sv = f[d];
    for (int k = 0; k < 25; ++k)
      atomicAdd(&urig[(size_t)a25[k] * 12 + d], ox[k % 5] * oy[k / 5] * sv);
  }
  // assemble MLP inputs and emit in A-fragment order
  float vals[64];
  float ph0 = phiag[row * 2], ph1 = phiag[row * 2 + 1];
  for (int c = 0; c < 12; ++c) {
    vals[c]      = f[c];
    vals[12 + c] = fsum[c];
    vals[26 + c] = Dij[c];
    vals[38 + c] = MU[c];
    vals[50 + c] = stag[(size_t)row * 12 + c];
  }
  vals[24] = ph0; vals[25] = ph1; vals[62] = 0.0f; vals[63] = 0.0f;
  store_row_frags(atr, row, vals);                       // feat_tr (62 -> 64)
  for (int c = 0; c < 12; ++c) vals[26 + c] = vals[50 + c];  // st_ag
  for (int c = 38; c < 64; ++c) vals[c] = 0.0f;
  store_row_frags(aq, row, vals);                        // feat_q (38 -> 64)
}

// ---------------------------------------------------------------------------
// WMMA helpers. All A/B operands live in fragment order: one contiguous v16h
// (32B) per lane per fragment -> b128-pair loads, no per-element packing.
// ---------------------------------------------------------------------------
DEVFN v16h frag_ld(const _Float16* base, int frag, int lane) {
  return *(const v16h*)(base + ((size_t)frag * 32 + lane) * 16);
}
// scatter a D-layout element (rowidx, col) into A-fragment-ordered LDS act
DEVFN void act_store(_Float16* act, int rowidx, int col, float x) {
  int kc = col >> 5, kcol = col & 31, g = kcol >> 3;
  int hl = g & 1, j = kcol - 8 * ((g + 1) >> 1);
  act[(size_t)kc * 512 + (hl * 16 + rowidx) * 16 + j] = (_Float16)x;
}
// one GEMM layer (relu+bias): A (frag-ordered, LDS or global), W (frag pool)
DEVFN void layer(const _Float16* Afrag, const _Float16* W, const float* bias,
                 int Kc, int Nt, _Float16* dstAct, int lane) {
  int hl = lane >> 4, m = lane & 15;
  v16h a0 = frag_ld(Afrag, 0, lane);
  v16h a1 = (Kc > 1) ? frag_ld(Afrag, 1, lane) : a0;
  for (int nt = 0; nt < Nt; ++nt) {
    v8f acc = {0.f, 0.f, 0.f, 0.f, 0.f, 0.f, 0.f, 0.f};
    v16h b0 = frag_ld(W, nt * Kc + 0, lane);
    acc = __builtin_amdgcn_wmma_f32_16x16x32_f16(false, a0, false, b0,
                                                 (short)0, acc, false, false);
    if (Kc > 1) {
      v16h b1 = frag_ld(W, nt * Kc + 1, lane);
      acc = __builtin_amdgcn_wmma_f32_16x16x32_f16(false, a1, false, b1,
                                                   (short)0, acc, false, false);
    }
    float bv = bias[nt * 16 + m];
#pragma unroll
    for (int r = 0; r < 8; ++r) {
      float x = fmaxf(acc[r] + bv, 0.0f);
      act_store(dstAct, r + 8 * hl, nt * 16 + m, x);
    }
  }
}
// single-output-column layer (N padded to 16, col 0); optional sigmoid
DEVFN void col_layer(const _Float16* Afrag, const _Float16* W,
                     const float* bias, int bidx, int Kc,
                     float* colbuf, float* gout, int rowBase, int t,
                     bool sigm, int lane) {
  v8f acc = {0.f, 0.f, 0.f, 0.f, 0.f, 0.f, 0.f, 0.f};
  v16h a0 = frag_ld(Afrag, 0, lane);
  v16h b0 = frag_ld(W, 0, lane);
  acc = __builtin_amdgcn_wmma_f32_16x16x32_f16(false, a0, false, b0,
                                               (short)0, acc, false, false);
  if (Kc > 1) {
    v16h a1 = frag_ld(Afrag, 1, lane);
    v16h b1 = frag_ld(W, 1, lane);
    acc = __builtin_amdgcn_wmma_f32_16x16x32_f16(false, a1, false, b1,
                                                 (short)0, acc, false, false);
  }
  int hl = lane >> 4, m = lane & 15;
  float b0s = bias[bidx];
  if (m == 0) {
#pragma unroll
    for (int r = 0; r < 8; ++r) {
      float x = acc[r] + b0s;
      if (sigm) x = 1.0f / (1.0f + __expf(-x));
      int mr = r + 8 * hl;
      if (colbuf) colbuf[mr] = x;
      gout[(size_t)(rowBase + mr) * Tc + t] = x;
    }
  }
}

struct K2Args {
  const _Float16* atr; const _Float16* aq; const _Float16* wp;
  const float *tb0, *tb1, *tb2, *qb0, *qb1, *qb2, *db0, *db1, *db2, *db3;
  const float* featst;
  float* featb; const float* udiff; const float* urig;
  float* out_us; float* out_mix; float* out_q0; float* out_fd;
  int t;
};

// ---------------------------------------------------------------------------
// K2: fused trans/q/dec MLPs + state update, one wave per 16-row tile
// ---------------------------------------------------------------------------
__global__ __launch_bounds__(128) void k_step2(K2Args A) {
  __shared__ __attribute__((aligned(32))) _Float16 sA[4][1024];
  __shared__ __attribute__((aligned(32))) _Float16 sB[4][1024];
  __shared__ float colb[4][16];
  int warp = threadIdx.x >> 5, lane = threadIdx.x & 31;
  int tile = blockIdx.x * 4 + warp;
  int rowBase = tile * 16;
  _Float16* a1 = &sA[warp][0];
  _Float16* a2 = &sB[warp][0];
  const _Float16* wp = A.wp;

  // ---- trans MLP: 62(64) -> 64 -> 32 -> mix = sigmoid(col 25) ----
  layer(A.atr + (size_t)tile * 1024, wp + O_TR0, A.tb0, 2, 4, a1, lane);
  __syncthreads();
  layer(a1, wp + O_TR1, A.tb1, 2, 2, a2, lane);
  __syncthreads();
  col_layer(a2, wp + O_TR2, A.tb2, 25, 1, colb[warp], A.out_mix, rowBase, A.t, true, lane);
  __syncthreads();

  // ---- q MLP: 38(64) -> 64 -> 32 -> 12 ----
  layer(A.aq + (size_t)tile * 1024, wp + O_Q0, A.qb0, 2, 4, a1, lane);
  __syncthreads();
  layer(a1, wp + O_Q1, A.qb1, 2, 2, a2, lane);
  __syncthreads();
  {
    // q final layer + state combine, dec input assembled into a1 (kc=0 block)
    v16h av = frag_ld(a2, 0, lane);
    v16h bv = frag_ld(wp + O_Q2, 0, lane);
    v8f acc = {0.f, 0.f, 0.f, 0.f, 0.f, 0.f, 0.f, 0.f};
    acc = __builtin_amdgcn_wmma_f32_16x16x32_f16(false, av, false, bv,
                                                 (short)0, acc, false, false);
    int hl = lane >> 4, m = lane & 15;
    int c = m;
    if (c < 12) {
      float bq = A.qb2[c];
#pragma unroll
      for (int r = 0; r < 8; ++r) {
        int mr = r + 8 * hl, row = rowBase + mr;
        float q0    = acc[r] + bq;
        float lastv = A.featb[(size_t)row * 12 + c];
        float ud    = A.udiff[(size_t)row * 12 + c];
        float ur    = A.urig[(size_t)row * 12 + c] - lastv;
        float mx    = colb[warp][mr];
        float nv    = lastv + mx * ud + (1.0f - mx) * ur + q0;
        A.featb[(size_t)row * 12 + c] = nv;
        A.out_q0[((size_t)row * 12 + c) * Tc + A.t] = q0;
        A.out_fd[((size_t)row * Tc + A.t) * 12 + c] = nv;
        act_store(a1, mr, c, nv);
      }
    }
#pragma unroll
    for (int r = 0; r < 8; ++r) {          // dec input cols 12..31: feat_st + zero pad
      int mr = r + 8 * hl, row = rowBase + mr;
      if (m >= 12)
        act_store(a1, mr, m, (float)A.featst[(size_t)row * 6 + (m - 12)]);
      int c2 = m + 16;
      act_store(a1, mr, c2, (c2 < 18) ? (float)A.featst[(size_t)row * 6 + (c2 - 12)] : 0.0f);
    }
  }
  __syncthreads();

  // ---- dec MLP: 18(32) -> 32 -> 32 -> 64 -> 1 ----
  layer(a1, wp + O_D0, A.db0, 1, 2, a2, lane); __syncthreads();
  layer(a2, wp + O_D1, A.db1, 1, 2, a1, lane); __syncthreads();
  layer(a1, wp + O_D2, A.db2, 1, 4, a2, lane); __syncthreads();
  col_layer(a2, wp + O_D3, A.db3, 0, 2, nullptr, A.out_us, rowBase, A.t, false, lane);
}

// ---------------------------------------------------------------------------
extern "C" void kernel_launch(void* const* d_in, const int* in_sizes, int n_in,
                              void* d_out, int out_size, void* d_ws, size_t ws_size,
                              hipStream_t stream) {
  (void)in_sizes; (void)n_in; (void)out_size; (void)ws_size;
  const float* phi     = (const float*)d_in[0];
  const float* feat_dy = (const float*)d_in[1];
  const float* feat_st = (const float*)d_in[2];
  const float* D       = (const float*)d_in[3];
  const float* mu      = (const float*)d_in[4];
  const float* cv      = (const float*)d_in[5];
  const int*   adj4    = (const int*)d_in[6];
  const int*   adj25   = (const int*)d_in[7];
  const float *tw0 = (const float*)d_in[8],  *tb0 = (const float*)d_in[9];
  const float *tw1 = (const float*)d_in[10], *tb1 = (const float*)d_in[11];
  const float *tw2 = (const float*)d_in[12], *tb2 = (const float*)d_in[13];
  const float *qw0 = (const float*)d_in[14], *qb0 = (const float*)d_in[15];
  const float *qw1 = (const float*)d_in[16], *qb1 = (const float*)d_in[17];
  const float *qw2 = (const float*)d_in[18], *qb2 = (const float*)d_in[19];
  const float *dw0 = (const float*)d_in[20], *db0 = (const float*)d_in[21];
  const float *dw1 = (const float*)d_in[22], *db1 = (const float*)d_in[23];
  const float *dw2 = (const float*)d_in[24], *db2 = (const float*)d_in[25];
  const float *dw3 = (const float*)d_in[26], *db3 = (const float*)d_in[27];

  char* w = (char*)d_ws;
  auto carve = [&](size_t bytes) -> void* {
    void* p = (void*)w;
    w += (bytes + 255) & ~(size_t)255;
    return p;
  };
  float* featb = (float*)carve((size_t)ROWS * 12 * sizeof(float));
  float* urig  = (float*)carve((size_t)ROWS * 12 * sizeof(float));
  float* udiff = (float*)carve((size_t)ROWS * 12 * sizeof(float));
  float* gpx   = (float*)carve((size_t)ROWS * sizeof(float));
  float* gpy   = (float*)carve((size_t)ROWS * sizeof(float));
  float* phiag = (float*)carve((size_t)ROWS * 2 * sizeof(float));
  float* stag  = (float*)carve((size_t)ROWS * 12 * sizeof(float));
  float* params = (float*)carve(256 * sizeof(float));
  _Float16* wpool = (_Float16*)carve((size_t)W_TOTAL * sizeof(_Float16));
  _Float16* atr   = (_Float16*)carve((size_t)ROWS * 64 * sizeof(_Float16));
  _Float16* aq    = (_Float16*)carve((size_t)ROWS * 64 * sizeof(_Float16));

  float* out_us  = (float*)d_out;
  float* out_mix = out_us  + (size_t)ROWS * Tc;
  float* out_q0  = out_mix + (size_t)ROWS * Tc;
  float* out_fd  = out_q0  + (size_t)ROWS * 12 * Tc;

  k_param<<<1, 256, 0, stream>>>(D, mu, cv, tw0, tw1, tw2, qw0, qw1, qw2,
                                 dw0, dw1, dw2, dw3, params, wpool);
  k_node<<<(ROWS + 255) / 256, 256, 0, stream>>>(phi, feat_dy, feat_st, adj4, adj25,
                                                 featb, gpx, gpy, phiag, stag);

  K2Args a2a;
  a2a.atr = atr; a2a.aq = aq; a2a.wp = wpool;
  a2a.tb0 = tb0; a2a.tb1 = tb1; a2a.tb2 = tb2;
  a2a.qb0 = qb0; a2a.qb1 = qb1; a2a.qb2 = qb2;
  a2a.db0 = db0; a2a.db1 = db1; a2a.db2 = db2; a2a.db3 = db3;
  a2a.featst = feat_st;
  a2a.featb = featb; a2a.udiff = udiff; a2a.urig = urig;
  a2a.out_us = out_us; a2a.out_mix = out_mix; a2a.out_q0 = out_q0; a2a.out_fd = out_fd;

  for (int t = 0; t < Tc; ++t) {
    k_zero<<<(ROWS * 12 + 255) / 256, 256, 0, stream>>>(urig, ROWS * 12);
    k_step1<<<(ROWS + 255) / 256, 256, 0, stream>>>(featb, gpx, gpy, phiag, stag,
                                                    params, adj4, adj25,
                                                    udiff, urig, atr, aq);
    a2a.t = t;
    k_step2<<<ROWS / 16 / 4, 128, 0, stream>>>(a2a);
  }
}
